// MultiheadAttention_86277303042557
// MI455X (gfx1250) — compile-verified, matrix operand order checked by reading, not run on hardware
//
#include <hip/hip_runtime.h>

#define NUM_HEADS 16
#define HEAD_SIZE 64
#define D_MODEL   1024
#define BATCH     2
#define SEQ       2048

typedef __attribute__((ext_vector_type(16))) _Float16 v16h;
typedef __attribute__((ext_vector_type(8)))  _Float16 v8h;
typedef __attribute__((ext_vector_type(4)))  _Float16 v4h;
typedef __attribute__((ext_vector_type(8)))  float    v8f;
typedef __attribute__((ext_vector_type(4)))  float    v4f;
typedef __attribute__((ext_vector_type(4)))  int      v4i;

// ---- CDNA5 async global->LDS path (guarded; falls back to bounce copy) ----
#if defined(__AMDGCN__) && defined(__gfx1250__) &&                            \
    __has_builtin(__builtin_amdgcn_global_load_async_to_lds_b128) &&          \
    __has_builtin(__builtin_amdgcn_s_wait_asynccnt)
#define USE_ASYNC_LDS 1
#else
#define USE_ASYNC_LDS 0
#endif

__device__ __forceinline__ void stage_b128(const _Float16* g, _Float16* l) {
#if USE_ASYNC_LDS
  __builtin_amdgcn_global_load_async_to_lds_b128(
      (__attribute__((address_space(1))) v4i*)(const_cast<_Float16*>(g)),
      (__attribute__((address_space(3))) v4i*)(l), 0, 0);
#else
  *(v8h*)l = *(const v8h*)g;
#endif
}

__device__ __forceinline__ void wait_stage() {
#if USE_ASYNC_LDS
  __builtin_amdgcn_s_wait_asynccnt(0);
#endif
}

// Load one WMMA f16 A/B fragment for a 32-wide K chunk.
// Caller passes p = row_base + k0 + ((lane>>4)<<3). Element j holds
// k = (j<8?0:16) + ((lane>>4)<<3) + (j&7)  ==> two contiguous 16B groups.
__device__ __forceinline__ v16h frag_load(const _Float16* p) {
  v8h lo = *(const v8h*)(p);
  v8h hi = *(const v8h*)(p + 16);
  v16h f;
#pragma unroll
  for (int i = 0; i < 8; ++i) { f[i] = lo[i]; f[i + 8] = hi[i]; }
  return f;
}

__device__ __forceinline__ v8f wmma_f16(v16h a, v16h b, v8f c) {
  return __builtin_amdgcn_wmma_f32_16x16x32_f16(false, a, false, b, (short)0, c,
                                                false, false);
}

// ---------------------------------------------------------------- convert
__global__ void cvt_f32_to_f16(const float* __restrict__ src,
                               _Float16* __restrict__ dst, int n4) {
  int i = blockIdx.x * blockDim.x + threadIdx.x;
  if (i >= n4) return;
  v4f x = ((const v4f*)src)[i];
  v4h h;
#pragma unroll
  for (int j = 0; j < 4; ++j) h[j] = (_Float16)x[j];
  ((v4h*)dst)[i] = h;
}

// ---------------------------------------------------------------- GEMM
// out[m,n] = sum_k A[m,k] * W[n,k]     (A:[M,K] f16, W:[N,K] f16)
// Block: 256 threads = 8 waves, each wave 32x64 -> block tile 256x64.
// W tile (64 rows x 32 k) staged in LDS via async DMA (double buffered,
// shared by all 8 waves); A fragments double-buffered in registers.
// mode 0: f16 [B,H,S,Dh]   mode 1: f16 [B,H,Dh,S]   mode 2: f32 [M,N]+bias
#define BK 32
#define BROW_PAD 40   // 80B row stride: 16B aligned, bank-friendly
__global__ void gemm_wmma(const _Float16* __restrict__ A,
                          const _Float16* __restrict__ W,
                          int M, int N, int K,
                          _Float16* __restrict__ outH,
                          float* __restrict__ outF,
                          const float* __restrict__ bias, int mode) {
  __shared__ __align__(16) _Float16 Bt[2][64][BROW_PAD];

  const int tid  = threadIdx.x;
  const int lane = tid & 31;
  const int wv   = tid >> 5;
  const int l16  = lane & 15;
  const int b8   = (lane >> 4) << 3;
  const int n0   = blockIdx.x * 64;
  const int m0   = blockIdx.y * 256 + wv * 32;

  // staging: each of 256 threads copies one 16B chunk of the 64x32 W tile
  const int srow = tid >> 2;          // 0..63
  const int scol = (tid & 3) * 8;     // 0,8,16,24 halfs
  const _Float16* gstage = W + (size_t)(n0 + srow) * K + scol;
  _Float16* lbuf[2] = {&Bt[0][srow][scol], &Bt[1][srow][scol]};

  v8f acc[2][4];
#pragma unroll
  for (int r = 0; r < 2; ++r)
#pragma unroll
    for (int t = 0; t < 4; ++t)
#pragma unroll
      for (int v = 0; v < 8; ++v) acc[r][t][v] = 0.0f;

  const _Float16* a0 = A + (size_t)(m0 + l16) * K + b8;
  const _Float16* a1 = A + (size_t)(m0 + 16 + l16) * K + b8;

  // prologue: stage first W tile, preload first A fragments
  stage_b128(gstage, lbuf[0]);
  v16h af0 = frag_load(a0);
  v16h af1 = frag_load(a1);
  wait_stage();
  __syncthreads();

  const int NS = K / BK;
  for (int s = 0; s < NS - 1; ++s) {
    const int k0 = s * BK;
    // overlap: next W tile via async DMA, next A fragments into registers
    stage_b128(gstage + k0 + BK, lbuf[(s + 1) & 1]);
    v16h an0 = frag_load(a0 + k0 + BK);
    v16h an1 = frag_load(a1 + k0 + BK);

    const _Float16* bbase = &Bt[s & 1][0][0];
#pragma unroll
    for (int t = 0; t < 4; ++t) {
      v16h bf = frag_load(bbase + (size_t)(t * 16 + l16) * BROW_PAD + b8);
      acc[0][t] = wmma_f16(af0, bf, acc[0][t]);
      acc[1][t] = wmma_f16(af1, bf, acc[1][t]);
    }
    af0 = an0;
    af1 = an1;
    wait_stage();
    __syncthreads();
  }
  {  // peeled last K-step: no staging, no barrier
    const _Float16* bbase = &Bt[(NS - 1) & 1][0][0];
#pragma unroll
    for (int t = 0; t < 4; ++t) {
      v16h bf = frag_load(bbase + (size_t)(t * 16 + l16) * BROW_PAD + b8);
      acc[0][t] = wmma_f16(af0, bf, acc[0][t]);
      acc[1][t] = wmma_f16(af1, bf, acc[1][t]);
    }
  }

#pragma unroll
  for (int r = 0; r < 2; ++r) {
#pragma unroll
    for (int t = 0; t < 4; ++t) {
#pragma unroll
      for (int v = 0; v < 8; ++v) {
        int m = m0 + r * 16 + v + b8;     // C layout: row = vgpr + 8*hi
        int n = n0 + t * 16 + l16;        //           col = lane & 15
        float val = acc[r][t][v];
        if (mode == 2) {
          outF[(size_t)m * N + n] = val + bias[n];
        } else {
          int b  = m / SEQ,       sI = m % SEQ;
          int h  = n / HEAD_SIZE, dh = n % HEAD_SIZE;
          size_t idx = (mode == 0)
            ? (((size_t)(b * NUM_HEADS + h) * SEQ + sI) * HEAD_SIZE + dh)
            : (((size_t)(b * NUM_HEADS + h) * HEAD_SIZE + dh) * SEQ + sI);
          outH[idx] = (_Float16)val;
        }
      }
    }
  }
}

// ---------------------------------------------------------------- attention
// Q,K: [B,H,S,Dh] f16   Vt: [B,H,Dh,S] f16   Oc: [B,S,D] f16 (concat layout)
// One wave per 16-row query tile; online-softmax over 32-key blocks.
// No block barriers: waves have different causal trip counts.
__global__ void flash_attn(const _Float16* __restrict__ Q,
                           const _Float16* __restrict__ Km,
                           const _Float16* __restrict__ Vt,
                           _Float16* __restrict__ Oc) {
  __shared__ __align__(16) _Float16 Plds[8][16][32];  // per-wave P staging

  const int lane = threadIdx.x & 31;
  const int wv   = threadIdx.x >> 5;
  const int l16  = lane & 15;
  const int b8   = (lane >> 4) << 3;
  const int bh   = blockIdx.y;                      // b*H + h
  const int q0   = (blockIdx.x * 8 + wv) * 16;

  const _Float16* Qp = Q  + (size_t)bh * SEQ * HEAD_SIZE;
  const _Float16* Kp = Km + (size_t)bh * SEQ * HEAD_SIZE;
  const _Float16* Vp = Vt + (size_t)bh * HEAD_SIZE * SEQ;

  v16h qf0 = frag_load(Qp + (size_t)(q0 + l16) * HEAD_SIZE + b8);
  v16h qf1 = frag_load(Qp + (size_t)(q0 + l16) * HEAD_SIZE + 32 + b8);

  v8f acc[4];
#pragma unroll
  for (int t = 0; t < 4; ++t)
#pragma unroll
    for (int v = 0; v < 8; ++v) acc[t][v] = 0.0f;
  float mrow[8], lrow[8];
#pragma unroll
  for (int v = 0; v < 8; ++v) { mrow[v] = -3.0e38f; lrow[v] = 0.0f; }

  const float scale = 0.125f;  // 1/sqrt(64)

  for (int k0 = 0; k0 < q0 + 16; k0 += 32) {
    // ---- scores: 16x32 tile = two 16x16 C tiles, K-dim = Dh = 64
    v8f sc[2];
#pragma unroll
    for (int t = 0; t < 2; ++t)
#pragma unroll
      for (int v = 0; v < 8; ++v) sc[t][v] = 0.0f;
#pragma unroll
    for (int t = 0; t < 2; ++t) {
      const _Float16* kr = Kp + (size_t)(k0 + t * 16 + l16) * HEAD_SIZE + b8;
      sc[t] = wmma_f16(qf0, frag_load(kr),      sc[t]);
      sc[t] = wmma_f16(qf1, frag_load(kr + 32), sc[t]);
    }

    // ---- scale, causal mask, online softmax (rows live in 16-lane groups)
#pragma unroll
    for (int v = 0; v < 8; ++v) {
      int m = q0 + v + b8;
      float rm = mrow[v];
#pragma unroll
      for (int t = 0; t < 2; ++t) {
        int n = k0 + t * 16 + l16;
        float val = sc[t][v] * scale;
        if (n > m) val = -3.0e38f;
        sc[t][v] = val;
        rm = fmaxf(rm, val);
      }
#pragma unroll
      for (int d = 1; d < 16; d <<= 1)
        rm = fmaxf(rm, __shfl_xor(rm, d, 32));
      float corr = __expf(mrow[v] - rm);
      float rsum = 0.0f;
#pragma unroll
      for (int t = 0; t < 2; ++t) {
        float p = __expf(sc[t][v] - rm);
        sc[t][v] = p;
        rsum += p;
      }
#pragma unroll
      for (int d = 1; d < 16; d <<= 1)
        rsum += __shfl_xor(rsum, d, 32);
      lrow[v] = lrow[v] * corr + rsum;
      mrow[v] = rm;
#pragma unroll
      for (int t = 0; t < 4; ++t) acc[t][v] *= corr;
      // stage P (C layout -> LDS row-major); same-wave DS ops are in-order
      Plds[wv][v + b8][l16]      = (_Float16)sc[0][v];
      Plds[wv][v + b8][16 + l16] = (_Float16)sc[1][v];
    }

    // ---- PV: A = P (16x32), B = Vt rows (contiguous over key positions)
    v16h pf = frag_load(&Plds[wv][l16][b8]);
#pragma unroll
    for (int t = 0; t < 4; ++t) {
      v16h vb = frag_load(Vp + (size_t)(t * 16 + l16) * SEQ + k0 + b8);
      acc[t] = wmma_f16(pf, vb, acc[t]);
    }
  }

  // ---- normalize + write concat [B,S,H*Dh]
  int b = bh / NUM_HEADS, h = bh % NUM_HEADS;
#pragma unroll
  for (int v = 0; v < 8; ++v) {
    float inv = 1.0f / lrow[v];
    int s = q0 + v + b8;
    size_t rowoff = ((size_t)b * SEQ + s) * D_MODEL + (size_t)h * HEAD_SIZE;
#pragma unroll
    for (int t = 0; t < 4; ++t)
      Oc[rowoff + t * 16 + l16] = (_Float16)(acc[t][v] * inv);
  }
}

// ---------------------------------------------------------------- launch
extern "C" void kernel_launch(void* const* d_in, const int* in_sizes, int n_in,
                              void* d_out, int out_size, void* d_ws,
                              size_t ws_size, hipStream_t stream) {
  (void)in_sizes; (void)n_in; (void)out_size; (void)ws_size;

  const float* query = (const float*)d_in[0];
  const float* key   = (const float*)d_in[1];
  const float* value = (const float*)d_in[2];
  const float* W_q   = (const float*)d_in[3];
  const float* W_k   = (const float*)d_in[4];
  const float* W_v   = (const float*)d_in[5];
  const float* W_ff  = (const float*)d_in[6];
  const float* b_ff  = (const float*)d_in[7];

  const size_t BSD = (size_t)BATCH * SEQ * D_MODEL;     // 4,194,304
  const size_t DD  = (size_t)D_MODEL * D_MODEL;         // 1,048,576
  _Float16* w    = (_Float16*)d_ws;
  _Float16* qh   = w;                 // fp16 inputs
  _Float16* kh   = w + BSD;
  _Float16* vh   = w + 2 * BSD;
  _Float16* wqh  = w + 3 * BSD;       // fp16 weights
  _Float16* wkh  = wqh + DD;
  _Float16* wvh  = wkh + DD;
  _Float16* wffh = wvh + DD;
  _Float16* Qh   = wffh + DD;         // projected heads
  _Float16* Kh   = Qh + BSD;
  _Float16* Vth  = Kh + BSD;
  _Float16* conc = qh;                // reuse: qh consumed before attention

  const int M = BATCH * SEQ;

  auto cvt = [&](const float* s, _Float16* d, size_t n) {
    int n4 = (int)(n / 4);
    cvt_f32_to_f16<<<dim3((n4 + 255) / 256), dim3(256), 0, stream>>>(s, d, n4);
  };
  cvt(query, qh, BSD);
  cvt(key,   kh, BSD);
  cvt(value, vh, BSD);
  cvt(W_q,  wqh, DD);
  cvt(W_k,  wkh, DD);
  cvt(W_v,  wvh, DD);
  cvt(W_ff, wffh, DD);

  dim3 gg(D_MODEL / 64, M / 256), gb(256);
  gemm_wmma<<<gg, gb, 0, stream>>>(qh, wqh, M, D_MODEL, D_MODEL,
                                   Qh, nullptr, nullptr, 0);
  gemm_wmma<<<gg, gb, 0, stream>>>(kh, wkh, M, D_MODEL, D_MODEL,
                                   Kh, nullptr, nullptr, 0);
  gemm_wmma<<<gg, gb, 0, stream>>>(vh, wvh, M, D_MODEL, D_MODEL,
                                   Vth, nullptr, nullptr, 1);

  flash_attn<<<dim3(SEQ / 128, BATCH * NUM_HEADS), dim3(256), 0, stream>>>(
      Qh, Kh, Vth, conc);

  gemm_wmma<<<gg, gb, 0, stream>>>(conc, wffh, M, D_MODEL, D_MODEL,
                                   nullptr, (float*)d_out, b_ff, 2);
}
